// MultiHeadSelfAttention_44246753083463
// MI455X (gfx1250) — matrix operand order repaired and checked by reading, not compiled, not run on hardware
//
#include <hip/hip_runtime.h>

#define D_MODEL 1024
#define NUM_HEADS 16
#define DK 64
#define SEQ 2048
#define BATCH 2
#define BS (BATCH * SEQ) /* 4096 */

typedef __attribute__((ext_vector_type(16))) _Float16     v16h;
typedef __attribute__((ext_vector_type(8)))  _Float16     v8h;
typedef __attribute__((ext_vector_type(8)))  float        v8f;
typedef __attribute__((ext_vector_type(4)))  unsigned int u32x4;
typedef __attribute__((ext_vector_type(8)))  int          i32x8;
typedef __attribute__((ext_vector_type(4)))  int          i32x4;

static __device__ __forceinline__ v8f wmma_f16(v16h a, v16h b, v8f c) {
    // D = A(16x32 f16) * B(32x16 f16) + C(16x16 f32)
    return __builtin_amdgcn_wmma_f32_16x16x32_f16(
        false, a, false, b, (short)0, c, false, false);
}

// A-matrix 16x32 f16 fragment (own k order, not the ISA table's): lane L holds row m = L%16;
// halves 0..15 -> K = (L/16)*16 + 0..15 (contiguous, matches B fragment)
static __device__ __forceinline__ v16h load_a_frag(const _Float16* base, int pitch, int lane) {
    int m  = lane & 15;
    int ko = (lane >> 4) * 16;
    const _Float16* p = base + (size_t)m * pitch + ko;
    union { v16h v; v8h h[2]; } u;
    u.h[0] = *(const v8h*)(p);
    u.h[1] = *(const v8h*)(p + 8);
    return u.v;
}

// B-matrix 32x16 f16 fragment: lane L holds column n = L%16 across rows
// K = (L/16)*16 .. +15, contiguous because sources are stored [N, K] row-major.
static __device__ __forceinline__ v16h load_b_frag(const _Float16* base, int pitch, int lane) {
    int n  = lane & 15;
    int kk = (lane >> 4) * 16;
    return *(const v16h*)(base + (size_t)n * pitch + kk);
}

// ---------------------------------------------------------------------------
// Tensor Data Mover: 2D tile (global -> LDS) with optional LDS padding.
// Builds a D# per CDNA5 ISA and issues
// tensor_load_to_lds (6-arg clang-23 builtin). Tracked by TENSORcnt.
//   flags = group1 dword0: data_size=2B | pad_enable | pad_interval | pad_amount
// ---------------------------------------------------------------------------
static __device__ __forceinline__ void tdm_load_2d(
    unsigned int lds_off, const _Float16* gptr,
    unsigned int tile0, unsigned int tile1,
    unsigned int tdim0, unsigned int tdim1,
    unsigned long long stride0, unsigned int flags) {
    unsigned long long ga = (unsigned long long)(size_t)gptr;
    u32x4 g0;
    g0[0] = 1u;                                            // count=1, user D#
    g0[1] = lds_off;                                       // lds_addr
    g0[2] = (unsigned int)(ga & 0xffffffffu);              // global_addr[31:0]
    g0[3] = (unsigned int)((ga >> 32) & 0x01ffffffu) | (2u << 30);  // type=2
    i32x8 g1;
    g1[0] = (int)flags;                                    // mask=0, data_size=2B, pads
    g1[1] = (int)((tdim0 & 0xffffu) << 16);                // tensor_dim0 @ bits 79:48
    g1[2] = (int)(((tdim0 >> 16) & 0xffffu) | ((tdim1 & 0xffffu) << 16));
    g1[3] = (int)(((tdim1 >> 16) & 0xffffu) | ((tile0 & 0xffffu) << 16));
    g1[4] = (int)(tile1 & 0xffffu);                        // tile_dim1 (tile_dim2=0)
    g1[5] = (int)(stride0 & 0xffffffffu);                  // tensor_dim0_stride[31:0]
    g1[6] = (int)((stride0 >> 32) & 0xffffu);              // [47:32]; dim1_stride=0
    g1[7] = 0;
    i32x4 z4 = {};
    i32x8 z8 = {};
    __builtin_amdgcn_tensor_load_to_lds(g0, g1, z4, z4, z8, 0);
}

#define TDM_FLAGS_K (0x00010000u | (1u << 20) | (4u << 22) | (3u << 25)) /* pad 4dw / 32dw */
#define TDM_FLAGS_V (0x00010000u | (1u << 20) | (3u << 22) | (3u << 25)) /* pad 4dw / 16dw */
#define KPITCH 72  /* 64 + 8 halves from TDM padding: conflict-free b128 LDS reads */
#define VPITCH 40  /* 32 + 8 halves */
#define PPITCH 40  /* padded P tile pitch */

// ---------------------------------------------------------------------------
// Kernel 1: fp32 -> fp16 staging (x, Wq, Wk, Wv, Wo) into workspace
// ---------------------------------------------------------------------------
__global__ void cvt_kernel(const float* __restrict__ x,  const float* __restrict__ wq,
                           const float* __restrict__ wk, const float* __restrict__ wv,
                           const float* __restrict__ wo, _Float16* __restrict__ ws) {
    const size_t NX = (size_t)BS * D_MODEL;
    const size_t NW = (size_t)D_MODEL * D_MODEL;
    const size_t total = NX + 4 * NW;
    for (size_t i = (size_t)blockIdx.x * blockDim.x + threadIdx.x; i < total;
         i += (size_t)gridDim.x * blockDim.x) {
        float v;
        if      (i < NX)          v = x[i];
        else if (i < NX + NW)     v = wq[i - NX];
        else if (i < NX + 2 * NW) v = wk[i - NX - NW];
        else if (i < NX + 3 * NW) v = wv[i - NX - 2 * NW];
        else                      v = wo[i - NX - 3 * NW];
        ws[i] = (_Float16)v;
    }
}

// ---------------------------------------------------------------------------
// Kernel 2: one-shot RoPE cos/sin tables: [SEQ][32] each.
// ---------------------------------------------------------------------------
__global__ void rope_tab_kernel(float* __restrict__ cosT, float* __restrict__ sinT) {
    int idx = blockIdx.x * blockDim.x + threadIdx.x;
    if (idx >= SEQ * 32) return;
    int s = idx >> 5;
    int p = idx & 31;
    float freq = powf(10000.0f, -(float)p * (1.0f / 32.0f));
    float ang  = (float)s * freq;
    cosT[idx] = cosf(ang);
    sinT[idx] = sinf(ang);
}

// ---------------------------------------------------------------------------
// Kernel 3: QKV projection (X * W^T), 64x64 per wave (4x4 WMMA frags),
// fused RoPE table lookup + head-split stores. z = 0:Q, 1:K, 2:V.
// ---------------------------------------------------------------------------
__global__ __launch_bounds__(256) void qkv_rope_kernel(
    const _Float16* __restrict__ Xh,
    const _Float16* __restrict__ Wqh, const _Float16* __restrict__ Wkh,
    const _Float16* __restrict__ Wvh,
    const float* __restrict__ cosT, const float* __restrict__ sinT,
    _Float16* __restrict__ Qh, _Float16* __restrict__ Kh, _Float16* __restrict__ Vt) {
    const int lane = threadIdx.x & 31;
    const int wl   = blockIdx.x * 8 + (threadIdx.x >> 5);
    const int TPM  = (BS / 64) * (D_MODEL / 64);  // 1024 wave-tiles per matrix
    const int z    = wl / TPM;
    const int rem  = wl % TPM;
    const int rt   = rem / (D_MODEL / 64);
    const int ct   = rem % (D_MODEL / 64);        // == head index

    const _Float16* W = (z == 0) ? Wqh : (z == 1) ? Wkh : Wvh;
    const _Float16* Abase = Xh + (size_t)rt * 64 * D_MODEL;
    const _Float16* Bbase = W  + (size_t)ct * 64 * D_MODEL;

    v8f acc[4][4] = {};
    for (int kk = 0; kk < D_MODEL; kk += 32) {
        v16h aF[4];
        #pragma unroll
        for (int i = 0; i < 4; ++i)
            aF[i] = load_a_frag(Abase + (size_t)i * 16 * D_MODEL + kk, D_MODEL, lane);
        #pragma unroll
        for (int c = 0; c < 4; ++c) {
            v16h bF = load_b_frag(Bbase + (size_t)c * 16 * D_MODEL + kk, D_MODEL, lane);
            #pragma unroll
            for (int i = 0; i < 4; ++i)
                acc[i][c] = wmma_f16(aF[i], bF, acc[i][c]);
        }
    }

    const int n  = lane & 15;
    const int hh = ct;
    #pragma unroll
    for (int i = 0; i < 4; ++i) {
        const int rbase = rt * 64 + i * 16 + ((lane >> 4) << 3);
        #pragma unroll
        for (int c = 0; c < 4; ++c) {
            const int dd = c * 16 + n;
            const int p  = dd >> 1;
            #pragma unroll
            for (int j = 0; j < 8; ++j) {
                const int row = rbase + j;
                const int b   = row >> 11;
                const int s   = row & (SEQ - 1);
                float val = acc[i][c][j];
                if (z < 2) {
                    const float cs = cosT[s * 32 + p];
                    const float sn = sinT[s * 32 + p];
                    const float prt = __shfl_xor(val, 1, 32);
                    val = ((dd & 1) == 0) ? (val * cs - prt * sn)
                                          : (prt * sn + val * cs);
                }
                if (z == 0)
                    Qh[((size_t)(b * NUM_HEADS + hh) * SEQ + s) * DK + dd] = (_Float16)val;
                else if (z == 1)
                    Kh[((size_t)(b * NUM_HEADS + hh) * SEQ + s) * DK + dd] = (_Float16)val;
                else
                    Vt[((size_t)(b * NUM_HEADS + hh) * DK + dd) * SEQ + s] = (_Float16)val;
            }
        }
    }
}

// ---------------------------------------------------------------------------
// Kernel 4: causal flash attention. All 8 waves of a block share one (b,h);
// K/V 32-key tiles are staged to LDS once per block by the TDM,
// double-buffered and overlapped with WMMA compute.
// ---------------------------------------------------------------------------
__global__ __launch_bounds__(256) void attn_kernel(
    const _Float16* __restrict__ Qh, const _Float16* __restrict__ Kh,
    const _Float16* __restrict__ Vt, _Float16* __restrict__ Oh) {
    __shared__ _Float16 ldsK[2][32 * KPITCH];   // 32 keys x 64 d (padded)
    __shared__ _Float16 ldsV[2][64 * VPITCH];   // 64 d x 32 keys (padded)
    __shared__ _Float16 ldsP[8][16 * PPITCH];   // per-wave P tile

    const int lane = threadIdx.x & 31;
    const int w    = threadIdx.x >> 5;
    const int bh   = blockIdx.x >> 4;           // b*NUM_HEADS + h
    const int sblk = blockIdx.x & 15;
    const int qg0  = sblk * 128 + w * 16;

    const _Float16* Qbase = Qh + ((size_t)bh * SEQ + qg0) * DK;
    const _Float16* Kbase = Kh + (size_t)bh * SEQ * DK;
    const _Float16* Vbase = Vt + (size_t)bh * DK * SEQ;

    const v16h qa0 = load_a_frag(Qbase, DK, lane);       // d 0..31
    const v16h qa1 = load_a_frag(Qbase + 32, DK, lane);  // d 32..63

    v8f accO[4] = {};
    float mrun[8], lrun[8];
    #pragma unroll
    for (int j = 0; j < 8; ++j) { mrun[j] = -__builtin_inff(); lrun[j] = 0.0f; }

    const int n      = lane & 15;
    const int rbase  = (lane >> 4) << 3;
    const int nkb    = (qg0 + 16 + 31) >> 5;              // this wave's blocks
    const int nkbMax = (sblk * 128 + 112 + 16 + 31) >> 5; // block-uniform
    _Float16* pl = &ldsP[w][0];

    const unsigned int offK0 = (unsigned int)(size_t)(const void*)&ldsK[0][0];
    const unsigned int offK1 = (unsigned int)(size_t)(const void*)&ldsK[1][0];
    const unsigned int offV0 = (unsigned int)(size_t)(const void*)&ldsV[0][0];
    const unsigned int offV1 = (unsigned int)(size_t)(const void*)&ldsV[1][0];

    if (w == 0) {   // prologue DMA for block 0
        tdm_load_2d(offK0, Kbase, 2048, 1, (unsigned)(SEQ * DK), 1,
                    (unsigned long long)(SEQ * DK), TDM_FLAGS_K);
        tdm_load_2d(offV0, Vbase, 32, 64, (unsigned)SEQ, DK,
                    (unsigned long long)SEQ, TDM_FLAGS_V);
    }

    for (int kb = 0; kb < nkbMax; ++kb) {
        const int k0 = kb * 32;
        if (w == 0) {
            if (kb + 1 < nkbMax) {   // DMA next block into the other buffer
                const int kn = k0 + 32;
                const unsigned int oK = ((kb + 1) & 1) ? offK1 : offK0;
                const unsigned int oV = ((kb + 1) & 1) ? offV1 : offV0;
                tdm_load_2d(oK, Kbase + (size_t)kn * DK, 2048, 1,
                            (unsigned)(SEQ * DK), 1,
                            (unsigned long long)(SEQ * DK), TDM_FLAGS_K);
                tdm_load_2d(oV, Vbase + kn, 32, 64, (unsigned)SEQ, DK,
                            (unsigned long long)SEQ, TDM_FLAGS_V);
                __builtin_amdgcn_s_wait_tensorcnt(2);  // block kb's pair done
            } else {
                __builtin_amdgcn_s_wait_tensorcnt(0);
            }
        }
        __syncthreads();             // K/V tile for kb visible to all waves

        if (kb < nkb) {              // wave-uniform causal skip
            const _Float16* kT = &ldsK[kb & 1][0];
            const _Float16* vT = &ldsV[kb & 1][0];

            // ---- scores: Q (16x64) * K^T (64x32) -> two 16x16 f32 tiles
            v8f c0 = {}, c1 = {};
            c0 = wmma_f16(qa0, load_b_frag(kT,                    KPITCH, lane), c0);
            c0 = wmma_f16(qa1, load_b_frag(kT + 32,               KPITCH, lane), c0);
            c1 = wmma_f16(qa0, load_b_frag(kT + 16 * KPITCH,      KPITCH, lane), c1);
            c1 = wmma_f16(qa1, load_b_frag(kT + 16 * KPITCH + 32, KPITCH, lane), c1);

            // ---- online softmax
            #pragma unroll
            for (int j = 0; j < 8; ++j) {
                const int row = qg0 + rbase + j;
                float s0 = c0[j] * 0.125f;
                float s1 = c1[j] * 0.125f;
                if (k0 + n > row)      s0 = -__builtin_inff();
                if (k0 + 16 + n > row) s1 = -__builtin_inff();
                float tm = fmaxf(s0, s1);
                tm = fmaxf(tm, __shfl_xor(tm, 1, 32));
                tm = fmaxf(tm, __shfl_xor(tm, 2, 32));
                tm = fmaxf(tm, __shfl_xor(tm, 4, 32));
                tm = fmaxf(tm, __shfl_xor(tm, 8, 32));
                const float mnew  = fmaxf(mrun[j], tm);
                const float alpha = __expf(mrun[j] - mnew);
                const float p0 = __expf(s0 - mnew);
                const float p1 = __expf(s1 - mnew);
                float rs = p0 + p1;
                rs += __shfl_xor(rs, 1, 32);
                rs += __shfl_xor(rs, 2, 32);
                rs += __shfl_xor(rs, 4, 32);
                rs += __shfl_xor(rs, 8, 32);
                lrun[j] = lrun[j] * alpha + rs;
                mrun[j] = mnew;
                #pragma unroll
                for (int f = 0; f < 4; ++f) accO[f][j] *= alpha;
                const int m = rbase + j;
                pl[m * PPITCH + n]      = (_Float16)p0;
                pl[m * PPITCH + 16 + n] = (_Float16)p1;
            }

            // ---- P (16x32) * V (32x64); DS ops are in-order within a wave
            const v16h pa = load_a_frag(pl, PPITCH, lane);
            #pragma unroll
            for (int f = 0; f < 4; ++f) {
                v16h vb = load_b_frag(vT + f * 16 * VPITCH, VPITCH, lane);
                accO[f] = wmma_f16(pa, vb, accO[f]);
            }
        }
        __syncthreads();             // all done reading before next DMA overwrite
    }

    // ---- finalize + store merged-head layout [b, s, D]
    const int b  = bh >> 4;
    const int hh = bh & 15;
    #pragma unroll
    for (int j = 0; j < 8; ++j) {
        const float inv = 1.0f / lrun[j];
        const int row = qg0 + rbase + j;
        const size_t base = ((size_t)b * SEQ + row) * D_MODEL + hh * DK + n;
        #pragma unroll
        for (int f = 0; f < 4; ++f)
            Oh[base + f * 16] = (_Float16)(accO[f][j] * inv);
    }
}

// ---------------------------------------------------------------------------
// Kernel 5: output projection O * Wo^T -> fp32 d_out, 64x64 per wave
// ---------------------------------------------------------------------------
__global__ __launch_bounds__(256) void out_proj_kernel(
    const _Float16* __restrict__ Oh, const _Float16* __restrict__ Woh,
    float* __restrict__ out) {
    const int lane = threadIdx.x & 31;
    const int wl   = blockIdx.x * 8 + (threadIdx.x >> 5);
    const int rt   = wl / (D_MODEL / 64);
    const int ct   = wl % (D_MODEL / 64);
    const _Float16* Abase = Oh  + (size_t)rt * 64 * D_MODEL;
    const _Float16* Bbase = Woh + (size_t)ct * 64 * D_MODEL;

    v8f acc[4][4] = {};
    for (int kk = 0; kk < D_MODEL; kk += 32) {
        v16h aF[4];
        #pragma unroll
        for (int i = 0; i < 4; ++i)
            aF[i] = load_a_frag(Abase + (size_t)i * 16 * D_MODEL + kk, D_MODEL, lane);
        #pragma unroll
        for (int c = 0; c < 4; ++c) {
            v16h bF = load_b_frag(Bbase + (size_t)c * 16 * D_MODEL + kk, D_MODEL, lane);
            #pragma unroll
            for (int i = 0; i < 4; ++i)
                acc[i][c] = wmma_f16(aF[i], bF, acc[i][c]);
        }
    }

    const int n = lane & 15;
    #pragma unroll
    for (int i = 0; i < 4; ++i) {
        const int rbase = rt * 64 + i * 16 + ((lane >> 4) << 3);
        #pragma unroll
        for (int c = 0; c < 4; ++c) {
            const int col = ct * 64 + c * 16 + n;
            #pragma unroll
            for (int j = 0; j < 8; ++j)
                out[(size_t)(rbase + j) * D_MODEL + col] = acc[i][c][j];
        }
    }
}

// ---------------------------------------------------------------------------
extern "C" void kernel_launch(void* const* d_in, const int* in_sizes, int n_in,
                              void* d_out, int out_size, void* d_ws, size_t ws_size,
                              hipStream_t stream) {
    (void)in_sizes; (void)n_in; (void)out_size; (void)ws_size;
    const float* x  = (const float*)d_in[0];
    const float* Wq = (const float*)d_in[1];
    const float* Wk = (const float*)d_in[2];
    const float* Wv = (const float*)d_in[3];
    const float* Wo = (const float*)d_in[4];

    const size_t NX = (size_t)BS * D_MODEL;
    const size_t NW = (size_t)D_MODEL * D_MODEL;

    _Float16* ws  = (_Float16*)d_ws;
    _Float16* Xh  = ws;
    _Float16* Wqh = Xh  + NX;
    _Float16* Wkh = Wqh + NW;
    _Float16* Wvh = Wkh + NW;
    _Float16* Woh = Wvh + NW;
    _Float16* Qh  = Woh + NW;      // [B, h, S, dk]
    _Float16* Kh  = Qh  + NX;      // [B, h, S, dk]
    _Float16* Vt  = Kh  + NX;      // [B, h, dk, S]
    _Float16* Oh  = Vt  + NX;      // [B, S, D]
    float*    cosT = (float*)(Oh + NX);
    float*    sinT = cosT + SEQ * 32;

    cvt_kernel<<<2048, 256, 0, stream>>>(x, Wq, Wk, Wv, Wo, ws);
    rope_tab_kernel<<<(SEQ * 32 + 255) / 256, 256, 0, stream>>>(cosT, sinT);
    qkv_rope_kernel<<<384, 256, 0, stream>>>(Xh, Wqh, Wkh, Wvh, cosT, sinT, Qh, Kh, Vt);
    attn_kernel<<<512, 256, 0, stream>>>(Qh, Kh, Vt, Oh);
    out_proj_kernel<<<128, 256, 0, stream>>>(Oh, Woh, (float*)d_out);
}
